// encoderModel_28673201668179
// MI455X (gfx1250) — matrix operand order repaired and checked by reading, not compiled, not run on hardware
//
#include <hip/hip_runtime.h>
#include <hip/hip_bf16.h>

// ---------------------------------------------------------------------------
// Problem constants (from reference)
// ---------------------------------------------------------------------------
#define BB 4
#define TT 64
#define NN 400
#define FIN 64
#define FL 128
#define RTOT (BB*TT*NN)          /* 102400 rows for node-major tensors  */
#define BT   (BB*TT)             /* 256 graph batches                   */
#define BN   (BB*NN)             /* 1600 temporal sequences             */

typedef __attribute__((ext_vector_type(16))) __bf16 v16bf;
typedef __attribute__((ext_vector_type(8)))  __bf16 v8bf;
typedef __attribute__((ext_vector_type(8)))  float  v8f;
typedef __attribute__((ext_vector_type(4)))  int    v4i;

// CDNA5 async global->LDS copy path (guarded; falls back to vector ld/st)
#if defined(__gfx1250__) && __has_builtin(__builtin_amdgcn_global_load_async_to_lds_b128) && __has_builtin(__builtin_amdgcn_s_wait_asynccnt)
#define ASYNC_COPY 1
#else
#define ASYNC_COPY 0
#endif

#if ASYNC_COPY
__device__ __forceinline__ void async16(const __bf16* g, __bf16* l) {
  // (global src, lds dst, imm offset, imm cpol)
  __builtin_amdgcn_global_load_async_to_lds_b128((v4i*)g, (v4i*)l, 0, 0);
}
__device__ __forceinline__ void wait_async(int n) {
  if (n >= 4)      __builtin_amdgcn_s_wait_asynccnt(4);
  else if (n >= 2) __builtin_amdgcn_s_wait_asynccnt(2);
  else             __builtin_amdgcn_s_wait_asynccnt(0);
}
#endif

__device__ __forceinline__ __bf16 f2bf(float f) {
  union { float f; unsigned u; } x; x.f = f;
  unsigned r = (x.u + 0x7FFFu + ((x.u >> 16) & 1u)) >> 16;
  union { unsigned short s; __bf16 b; } y; y.s = (unsigned short)r;
  return y.b;
}
__device__ __forceinline__ __bf16 bfz() {
  union { unsigned short s; __bf16 b; } y; y.s = 0; return y.b;
}

__device__ __forceinline__ float apply_act(float v, int act) {
  if (act == 1) return fmaxf(v, 0.0f);
  if (act == 2) return 0.5f * v * (1.0f + erff(v * 0.70710678118654752f));
  return v;
}

__device__ __forceinline__ float blk_reduce_sum(float v, float* s) {
  int tid = threadIdx.x, bd = blockDim.x;
  s[tid] = v; __syncthreads();
  for (int st = bd >> 1; st > 0; st >>= 1) {
    if (tid < st) s[tid] += s[tid + st];
    __syncthreads();
  }
  float r = s[0]; __syncthreads();
  return r;
}

// ---------------------------------------------------------------------------
// Batched GEMM:  C = alpha * (A @ B) + beta * C  (+bias, +activation)
// A: bf16 [M x K] (lda). B: bf16 [K x Nc] (ldb) or [Nc x K] if transB.
// Cf (f32) and/or Cbf (bf16 mirror) outputs, same geometry.
// Batch z = (zo=z/innerN, zi=z%innerN), independent outer/inner strides.
// 64x64 block tile, 4 waves x 16 cols, WMMA bf16 16x16x32, f32 accumulate.
// Double-buffered LDS tiles; async global->LDS copies on the fast path.
// ---------------------------------------------------------------------------
__global__ void __launch_bounds__(128)
k_gemm_bf16(const __bf16* __restrict__ A, const __bf16* __restrict__ B,
            float* __restrict__ Cf, __bf16* __restrict__ Cbf,
            const float* __restrict__ bias,
            int M, int Nc, int K, int lda, int ldb, int ldc,
            long long sAo, long long sAi, long long sBo, long long sBi,
            long long sCo, long long sCi, int innerN, int transB,
            float alpha, float beta, int act)
{
  const int z  = blockIdx.z;
  const int zo = z / innerN, zi = z % innerN;
  const __bf16* Ab = A + (size_t)zo * sAo + (size_t)zi * sAi;
  const __bf16* Bb = B + (size_t)zo * sBo + (size_t)zi * sBi;
  const size_t cOff = (size_t)zo * sCo + (size_t)zi * sCi;

  const int rowBase = blockIdx.y * 64;
  const int colBase = blockIdx.x * 64;
  const int tid  = threadIdx.x;
  const int lane = tid & 31;
  const int wv   = tid >> 5;
  const int lHi  = lane >> 4;
  const int lLo  = lane & 15;

  __shared__ __align__(16) __bf16 As[2][64][48];   // [m][k], 96B row stride
  __shared__ __align__(16) __bf16 Bs[2][64][48];   // [n][k], k-contiguous

  const bool fullM = (rowBase + 64 <= M);
  const bool fullN = (colBase + 64 <= Nc);

  // ---- accumulators (optionally preloaded with (beta/alpha)*C) ----
  v8f acc[4];
  #pragma unroll
  for (int mi = 0; mi < 4; ++mi)
    #pragma unroll
    for (int r = 0; r < 8; ++r) acc[mi][r] = 0.0f;
  if (beta != 0.0f && Cf) {
    const float cscale = beta / alpha;
    #pragma unroll
    for (int mi = 0; mi < 4; ++mi) {
      #pragma unroll
      for (int r = 0; r < 8; ++r) {
        int row = rowBase + mi * 16 + lHi * 8 + r;
        int col = colBase + wv * 16 + lLo;
        if ((fullM && fullN) || (row < M && col < Nc))
          acc[mi][r] = Cf[cOff + (size_t)row * ldc + col] * cscale;
      }
    }
  }

  const int arow = tid >> 1;            // 0..63 : tile row (A) / tile col (Bt)
  const int acb  = (tid & 1) * 16;      // 0 / 16 : k-chunk
  const int bk   = tid >> 2;            // 0..31 : k row (non-trans B)
  const int bn0  = (tid & 3) * 16;      // n chunk (non-trans B)

  auto stage = [&](int buf, int k0) -> int {
    int issued = 0;
    const bool fullK = (k0 + 32 <= K);
    // ---- stage A tile As[buf][m][k] ----
    if (fullM && fullK) {
      const __bf16* src = Ab + (size_t)(rowBase + arow) * lda + k0 + acb;
      __bf16* dst = &As[buf][arow][acb];
#if ASYNC_COPY
      async16(src, dst); async16(src + 8, dst + 8); issued += 2;
#else
      *(v8bf*)dst = *(const v8bf*)src;
      *(v8bf*)(dst + 8) = *(const v8bf*)(src + 8);
#endif
    } else {
      int gr = rowBase + arow;
      #pragma unroll
      for (int j = 0; j < 16; ++j) {
        int gk = k0 + acb + j;
        As[buf][arow][acb + j] =
            (gr < M && gk < K) ? Ab[(size_t)gr * lda + gk] : bfz();
      }
    }
    // ---- stage B tile Bs[buf][n][k] ----
    if (transB) {
      if (fullN && fullK) {
        const __bf16* src = Bb + (size_t)(colBase + arow) * ldb + k0 + acb;
        __bf16* dst = &Bs[buf][arow][acb];
#if ASYNC_COPY
        async16(src, dst); async16(src + 8, dst + 8); issued += 2;
#else
        *(v8bf*)dst = *(const v8bf*)src;
        *(v8bf*)(dst + 8) = *(const v8bf*)(src + 8);
#endif
      } else {
        int gn = colBase + arow;
        #pragma unroll
        for (int j = 0; j < 16; ++j) {
          int gk = k0 + acb + j;
          Bs[buf][arow][acb + j] =
              (gn < Nc && gk < K) ? Bb[(size_t)gn * ldb + gk] : bfz();
        }
      }
    } else {
      if (fullN && fullK) {
        const __bf16* src = Bb + (size_t)(k0 + bk) * ldb + colBase + bn0;
        if (k0 + 32 < K) __builtin_prefetch(src + (size_t)32 * ldb, 0, 1);
        v8bf v0 = *(const v8bf*)src;
        v8bf v1 = *(const v8bf*)(src + 8);
        #pragma unroll
        for (int j = 0; j < 8; ++j) {
          Bs[buf][bn0 + j][bk]     = v0[j];
          Bs[buf][bn0 + 8 + j][bk] = v1[j];
        }
      } else {
        #pragma unroll
        for (int j = 0; j < 16; ++j) {
          int gn = colBase + bn0 + j;
          int gk = k0 + bk;
          Bs[buf][bn0 + j][bk] =
              (gn < Nc && gk < K) ? Bb[(size_t)gk * ldb + gn] : bfz();
        }
      }
    }
    return issued;
  };

  const int nsteps = (K + 31) >> 5;
  stage(0, 0);
  for (int s = 0; s < nsteps; ++s) {
    const int cur = s & 1;
    int nxt = 0;
    if (s + 1 < nsteps) nxt = stage(cur ^ 1, (s + 1) * 32);
#if ASYNC_COPY
    wait_async(nxt);   // all older (current-tile) async copies complete
#else
    (void)nxt;
#endif
    __syncthreads();   // publish current tile across waves

    // B fragment: lane -> N = wv*16+lLo ; K lHi*8..+8 & 16+lHi*8..+8 (matches A)
    union UB { v16bf v; __bf16 e[16]; } bfr;
    {
      const __bf16* bp = &Bs[cur][wv * 16 + lLo][lHi * 8];
      const __bf16* bq = &Bs[cur][wv * 16 + lLo][16 + lHi * 8];
      #pragma unroll
      for (int j = 0; j < 8; ++j) { bfr.e[j] = bp[j]; bfr.e[8 + j] = bq[j]; }
    }
#if defined(__gfx1250__)
    #pragma unroll
    for (int mi = 0; mi < 4; ++mi) {
      // A fragment (16-bit 16x32 ISA layout): lane m=lLo;
      // lanes 0-15: K 0..7 & 16..23 ; lanes 16-31: K 8..15 & 24..31
      union UA { v16bf v; __bf16 e[16]; } afr;
      const __bf16* a0 = &As[cur][mi * 16 + lLo][lHi * 8];
      const __bf16* a1 = &As[cur][mi * 16 + lLo][16 + lHi * 8];
      #pragma unroll
      for (int j = 0; j < 8; ++j) { afr.e[j] = a0[j]; afr.e[8 + j] = a1[j]; }
      acc[mi] = __builtin_amdgcn_wmma_f32_16x16x32_bf16(
          false, afr.v, false, bfr.v, (short)0, acc[mi], false, false);
    }
#else
    (void)bfr;  // host pass only
#endif
    __syncthreads();   // protect tiles before next staging overwrite
  }

  // ---- epilogue ----
  #pragma unroll
  for (int mi = 0; mi < 4; ++mi) {
    #pragma unroll
    for (int r = 0; r < 8; ++r) {
      int row = rowBase + mi * 16 + lHi * 8 + r;
      int col = colBase + wv * 16 + lLo;
      bool ok = (fullM && fullN) || (row < M && col < Nc);
      if (ok) {
        float v = alpha * acc[mi][r];
        if (bias) v += bias[col];
        v = apply_act(v, act);
        size_t idx = cOff + (size_t)row * ldc + col;
        if (Cf)  Cf[idx]  = v;
        if (Cbf) Cbf[idx] = f2bf(v);
      }
    }
  }
}

// ---------------------------------------------------------------------------
// Elementwise / reduction kernels
// ---------------------------------------------------------------------------
__global__ void k_f2bf(const float* __restrict__ s, __bf16* __restrict__ d,
                       long long n) {
  long long i = (long long)blockIdx.x * blockDim.x + threadIdx.x;
  if (i < n) d[i] = f2bf(s[i]);
}

// D_n = sum_m |A[n,m]| + 1 ; dis = rsqrt(D)
__global__ void k_rowsum_dis(const float* __restrict__ A, float* __restrict__ dis) {
  __shared__ float s[128];
  size_t row = blockIdx.x;
  const float* a = A + row * NN;
  float acc = 0.0f;
  for (int i = threadIdx.x; i < NN; i += blockDim.x) acc += fabsf(a[i]);
  float D = blk_reduce_sum(acc, s) + 1.0f;
  if (threadIdx.x == 0) dis[row] = (D > 0.0f) ? rsqrtf(D) : 0.0f;
}

// A_norm = (|A| + I) * dis[n] * dis[m]   (f32 + bf16 mirror)
__global__ void k_anorm(const float* __restrict__ A, const float* __restrict__ dis,
                        float* __restrict__ an, __bf16* __restrict__ anb,
                        long long total) {
  long long i = (long long)blockIdx.x * blockDim.x + threadIdx.x;
  if (i >= total) return;
  int m = (int)(i % NN); long long t = i / NN;
  int n = (int)(t % NN); long long bt = t / NN;
  float v = fabsf(A[i]) + ((n == m) ? 1.0f : 0.0f);
  float r = v * dis[bt * NN + n] * dis[bt * NN + m];
  an[i] = r; anb[i] = f2bf(r);
}

// per-row L2 norm over F=128
__global__ void k_rownorm(const float* __restrict__ x, float* __restrict__ out) {
  __shared__ float s[128];
  size_t row = blockIdx.x;
  float v = x[row * FL + threadIdx.x];
  float ss = blk_reduce_sum(v * v, s);
  if (threadIdx.x == 0) out[row] = sqrtf(ss);
}

// scores = A_norm * dot / max(qn*kn, 1e-8)
__global__ void k_cosmask(float* __restrict__ sc, const float* __restrict__ an,
                          const float* __restrict__ qn, const float* __restrict__ kn,
                          long long total) {
  long long i = (long long)blockIdx.x * blockDim.x + threadIdx.x;
  if (i >= total) return;
  int m = (int)(i % NN); long long t = i / NN;
  int n = (int)(t % NN); long long bt = t / NN;
  float d = fmaxf(qn[bt * NN + n] * kn[bt * NN + m], 1e-8f);
  sc[i] = an[i] * (sc[i] / d);
}

// row softmax over last dim L; optional bf16 mirror of the result
__global__ void k_softmax(float* __restrict__ x, __bf16* __restrict__ xb, int L) {
  __shared__ float s[256];
  size_t base = (size_t)blockIdx.x * L;
  int tid = threadIdx.x, bd = blockDim.x;
  float m = -3.4e38f;
  for (int i = tid; i < L; i += bd) m = fmaxf(m, x[base + i]);
  s[tid] = m; __syncthreads();
  for (int st = bd >> 1; st > 0; st >>= 1) {
    if (tid < st) s[tid] = fmaxf(s[tid], s[tid + st]);
    __syncthreads();
  }
  m = s[0]; __syncthreads();
  float sum = 0.0f;
  for (int i = tid; i < L; i += bd) {
    float e = expf(x[base + i] - m);
    x[base + i] = e; sum += e;
  }
  float tot = blk_reduce_sum(sum, s);
  float inv = 1.0f / tot;
  for (int i = tid; i < L; i += bd) {
    float v = x[base + i] * inv;
    x[base + i] = v;
    if (xb) xb[base + i] = f2bf(v);
  }
}

// h = LayerNorm( (mode? relu(y) : y) + h ) * g + b  (in place; bf16 mirror)
__global__ void k_res_ln(const float* __restrict__ y, float* __restrict__ h,
                         __bf16* __restrict__ hb,
                         const float* __restrict__ g, const float* __restrict__ b,
                         int mode) {
  __shared__ float s[128];
  size_t row = blockIdx.x; int f = threadIdx.x;
  float v = y[row * FL + f];
  if (mode == 1) v = fmaxf(v, 0.0f);
  v += h[row * FL + f];
  float mean = blk_reduce_sum(v, s) * (1.0f / FL);
  float d = v - mean;
  float var = blk_reduce_sum(d * d, s) * (1.0f / FL);
  float r = rsqrtf(var + 1e-5f);
  float o = d * r * g[f] + b[f];
  h[row * FL + f] = o;
  if (hb) hb[row * FL + f] = f2bf(o);
}

// x[bn,t,f] = 0.5*sp[b,t,n,f] + 0.5*che[b,t,n,f] + PE[t,f]  (+bf16 mirror)
__global__ void k_mix_pe(const float* __restrict__ sp, const float* __restrict__ che,
                         float* __restrict__ x, __bf16* __restrict__ xb,
                         long long total) {
  long long i = (long long)blockIdx.x * blockDim.x + threadIdx.x;
  if (i >= total) return;
  int f = (int)(i % FL); long long rw = i / FL;
  int t = (int)(rw % TT); long long bn = rw / TT;
  int n = (int)(bn % NN); int b = (int)(bn / NN);
  size_t src = (((size_t)b * TT + t) * NN + n) * FL + f;
  float div = expf(-9.2103403719761836f * (float)(2 * (f >> 1)) * (1.0f / FL));
  float ang = (float)t * div;
  float pe = (f & 1) ? cosf(ang) : sinf(ang);
  float v = 0.5f * sp[src] + 0.5f * che[src] + pe;
  x[i] = v; xb[i] = f2bf(v);
}

// out[b,t,n,f] = x[(b*N+n)*T+t, f]
__global__ void k_unpermute(const float* __restrict__ x, float* __restrict__ out,
                            long long total) {
  long long i = (long long)blockIdx.x * blockDim.x + threadIdx.x;
  if (i >= total) return;
  int f = (int)(i % FL); long long t1 = i / FL;
  int n = (int)(t1 % NN); long long t2 = t1 / NN;
  int t = (int)(t2 % TT); int b = (int)(t2 / TT);
  out[i] = x[(((size_t)b * NN + n) * TT + t) * FL + f];
}

// ---------------------------------------------------------------------------
// Host orchestration
// ---------------------------------------------------------------------------
static inline void gemm(hipStream_t st, const __bf16* A, const __bf16* B,
                        float* Cf, __bf16* Cbf, const float* bias,
                        int M, int Nc, int K, int lda, int ldb, int ldc,
                        long long sAo, long long sAi, long long sBo, long long sBi,
                        long long sCo, long long sCi, int innerN, int batches,
                        int transB, float alpha, float beta, int act) {
  dim3 g((Nc + 63) / 64, (M + 63) / 64, batches);
  k_gemm_bf16<<<g, dim3(128), 0, st>>>(A, B, Cf, Cbf, bias, M, Nc, K,
                                       lda, ldb, ldc, sAo, sAi, sBo, sBi,
                                       sCo, sCi, innerN, transB, alpha, beta, act);
}

extern "C" void kernel_launch(void* const* d_in, const int* in_sizes, int n_in,
                              void* d_out, int out_size, void* d_ws, size_t ws_size,
                              hipStream_t stream) {
  (void)in_sizes; (void)n_in; (void)out_size; (void)ws_size;
  #define IN(k) ((const float*)d_in[k])

  const float* X  = IN(0);
  const float* Am = IN(1);
  int i = 2;
  const float* che_in_b = IN(i); i++;
  const float* che_in_w = IN(i); i++;
  struct CheL { const float *W, *b, *g; } che[2];
  for (int l = 0; l < 2; ++l) { che[l].W = IN(i); i++; che[l].b = IN(i); i++; che[l].g = IN(i); i++; }
  const float* sp_in_b = IN(i); i++;
  const float* sp_in_w = IN(i); i++;
  struct SpL { const float *b, *bk, *bq, *bv, *g, *wk, *wq, *wv; } sp[2];
  for (int l = 0; l < 2; ++l) {
    sp[l].b = IN(i); i++; sp[l].bk = IN(i); i++; sp[l].bq = IN(i); i++; sp[l].bv = IN(i); i++;
    sp[l].g = IN(i); i++; sp[l].wk = IN(i); i++; sp[l].wq = IN(i); i++; sp[l].wv = IN(i); i++;
  }
  struct TrL { const float *b1, *b2, *bf1, *bf2, *bo, *bqkv, *g1, *g2, *w1, *w2, *wo, *wqkv; } tr[2];
  for (int l = 0; l < 2; ++l) {
    tr[l].b1 = IN(i); i++; tr[l].b2 = IN(i); i++; tr[l].bf1 = IN(i); i++; tr[l].bf2 = IN(i); i++;
    tr[l].bo = IN(i); i++; tr[l].bqkv = IN(i); i++; tr[l].g1 = IN(i); i++; tr[l].g2 = IN(i); i++;
    tr[l].w1 = IN(i); i++; tr[l].w2 = IN(i); i++; tr[l].wo = IN(i); i++; tr[l].wqkv = IN(i); i++;
  }

  // ---- workspace layout (byte allocator, 256B aligned) ----
  char* base = (char*)d_ws;
  size_t off = 0;
  auto takeF = [&](size_t n) { off = (off + 255) & ~(size_t)255; float* r = (float*)(base + off); off += n * 4; return r; };
  auto takeH = [&](size_t n) { off = (off + 255) & ~(size_t)255; __bf16* r = (__bf16*)(base + off); off += n * 2; return r; };
  const size_t R = RTOT;
  const size_t NN2 = (size_t)BT * NN * NN;      // 40,960,000

  float*  dis   = takeF(R);
  float*  anorm = takeF(NN2);
  __bf16* anbf  = takeH(NN2);
  float*  bufA  = takeF(R * FL);   __bf16* hA = takeH(R * FL);
  float*  bufB  = takeF(R * FL);   __bf16* hB = takeH(R * FL);
  float*  bufC  = takeF(R * FL);   __bf16* hC = takeH(R * FL);
  float*  bufD  = takeF(R * FL);   __bf16* hD = takeH(R * FL);
  float*  bufE  = takeF(R * FL);   __bf16* hE = takeH(R * FL);
  float*  bufF  = takeF(R * FL);
  float*  qn    = takeF(R);
  float*  kn    = takeF(R);
  float*  big   = takeF(R * 512);               // sp scores / t scores (f32)
  __bf16* bigbf = takeH(R * 512);               // softmax mirrors / ff hidden
  __bf16* qkvbf = takeH(R * 384);
  __bf16* Xbf   = takeH(R * FIN);
  // weight mirrors (bf16)
  __bf16* w_che_in = takeH(FIN * FL);
  __bf16* w_cheW[2]; for (int l = 0; l < 2; ++l) w_cheW[l] = takeH(3 * FL * FL);
  __bf16* w_sp_in  = takeH(FIN * FL);
  __bf16 *w_q[2], *w_k[2], *w_v[2];
  for (int l = 0; l < 2; ++l) { w_q[l] = takeH(FL * FL); w_k[l] = takeH(FL * FL); w_v[l] = takeH(FL * FL); }
  __bf16 *w_qkv[2], *w_o[2], *w_1[2], *w_2[2];
  for (int l = 0; l < 2; ++l) {
    w_qkv[l] = takeH(FL * 384); w_o[l] = takeH(FL * FL);
    w_1[l] = takeH(FL * 512);   w_2[l] = takeH(512 * FL);
  }

  auto cvt = [&](const float* s, __bf16* d, long long n) {
    k_f2bf<<<dim3((unsigned)((n + 255) / 256)), dim3(256), 0, stream>>>(s, d, n);
  };

  const long long NNTOT = (long long)NN2;           // 40,960,000
  const long long RF    = (long long)R * FL;        // 13,107,200

  // ---- one-time conversions (weights + X) ----
  cvt(X, Xbf, (long long)R * FIN);
  cvt(che_in_w, w_che_in, FIN * FL);
  cvt(sp_in_w,  w_sp_in,  FIN * FL);
  for (int l = 0; l < 2; ++l) {
    cvt(che[l].W, w_cheW[l], 3 * FL * FL);
    cvt(sp[l].wq, w_q[l], FL * FL);
    cvt(sp[l].wk, w_k[l], FL * FL);
    cvt(sp[l].wv, w_v[l], FL * FL);
    cvt(tr[l].wqkv, w_qkv[l], FL * 384);
    cvt(tr[l].wo,   w_o[l],   FL * FL);
    cvt(tr[l].w1,   w_1[l],   FL * 512);
    cvt(tr[l].w2,   w_2[l],   512 * FL);
  }

  // ---- adjacency normalization ----
  k_rowsum_dis<<<dim3(R), dim3(128), 0, stream>>>(Am, dis);
  k_anorm<<<dim3((unsigned)((NNTOT + 255) / 256)), dim3(256), 0, stream>>>(Am, dis, anorm, anbf, NNTOT);

  // ---- Chebyshev branch ----
  gemm(stream, Xbf, w_che_in, bufA, hA, che_in_b, RTOT, FL, FIN, FIN, FL, FL,
       0, 0, 0, 0, 0, 0, 1, 1, 0, 1.0f, 0.0f, 0);
  for (int l = 0; l < 2; ++l) {
    const __bf16* W0 = w_cheW[l];
    const __bf16* W1 = w_cheW[l] + FL * FL;
    const __bf16* W2 = w_cheW[l] + 2 * FL * FL;
    // Tx1 = L2 @ h = -(A_norm @ h)
    gemm(stream, anbf, hA, nullptr, hB, nullptr, NN, FL, NN, NN, FL, FL,
         (long long)NN * NN, 0, (long long)NN * FL, 0, (long long)NN * FL, 0,
         1, BT, 0, -1.0f, 0.0f, 0);
    // y = Tx0@W0 ; y += Tx1@W1
    gemm(stream, hA, W0, bufD, nullptr, nullptr, RTOT, FL, FL, FL, FL, FL,
         0, 0, 0, 0, 0, 0, 1, 1, 0, 1.0f, 0.0f, 0);
    gemm(stream, hB, W1, bufD, nullptr, nullptr, RTOT, FL, FL, FL, FL, FL,
         0, 0, 0, 0, 0, 0, 1, 1, 0, 1.0f, 1.0f, 0);
    // Tx2 = 2*L2@Tx1 - Tx0 == -2*(A_norm@Tx1) - Tx0 (preload C with Tx0)
    (void)hipMemcpyAsync(bufC, bufA, RF * sizeof(float), hipMemcpyDeviceToDevice, stream);
    gemm(stream, anbf, hB, bufC, hC, nullptr, NN, FL, NN, NN, FL, FL,
         (long long)NN * NN, 0, (long long)NN * FL, 0, (long long)NN * FL, 0,
         1, BT, 0, -2.0f, -1.0f, 0);
    // y += Tx2@W2
    gemm(stream, hC, W2, bufD, nullptr, nullptr, RTOT, FL, FL, FL, FL, FL,
         0, 0, 0, 0, 0, 0, 1, 1, 0, 1.0f, 1.0f, 0);
    // h = LN(relu(y) + h)
    k_res_ln<<<dim3(R), dim3(128), 0, stream>>>(bufD, bufA, hA, che[l].g, che[l].b, 1);
  }
  // che_out = bufA / hA

  // ---- Spatial attention branch ----
  gemm(stream, Xbf, w_sp_in, bufB, hB, sp_in_b, RTOT, FL, FIN, FIN, FL, FL,
       0, 0, 0, 0, 0, 0, 1, 1, 0, 1.0f, 0.0f, 0);
  for (int l = 0; l < 2; ++l) {
    gemm(stream, hB, w_q[l], bufC, hC, sp[l].bq, RTOT, FL, FL, FL, FL, FL,
         0, 0, 0, 0, 0, 0, 1, 1, 0, 1.0f, 0.0f, 0);
    gemm(stream, hB, w_k[l], bufD, hD, sp[l].bk, RTOT, FL, FL, FL, FL, FL,
         0, 0, 0, 0, 0, 0, 1, 1, 0, 1.0f, 0.0f, 0);
    gemm(stream, hB, w_v[l], nullptr, hE, sp[l].bv, RTOT, FL, FL, FL, FL, FL,
         0, 0, 0, 0, 0, 0, 1, 1, 0, 1.0f, 0.0f, 0);
    k_rownorm<<<dim3(R), dim3(128), 0, stream>>>(bufC, qn);
    k_rownorm<<<dim3(R), dim3(128), 0, stream>>>(bufD, kn);
    // dot = Q @ K^T per (b,t)
    gemm(stream, hC, hD, big, nullptr, nullptr, NN, NN, FL, FL, FL, NN,
         (long long)NN * FL, 0, (long long)NN * FL, 0, (long long)NN * NN, 0,
         1, BT, 1, 1.0f, 0.0f, 0);
    k_cosmask<<<dim3((unsigned)((NNTOT + 255) / 256)), dim3(256), 0, stream>>>(big, anorm, qn, kn, NNTOT);
    k_softmax<<<dim3(R), dim3(128), 0, stream>>>(big, bigbf, NN);
    // y = attn @ V
    gemm(stream, bigbf, hE, bufF, nullptr, nullptr, NN, FL, NN, NN, FL, FL,
         (long long)NN * NN, 0, (long long)NN * FL, 0, (long long)NN * FL, 0,
         1, BT, 0, 1.0f, 0.0f, 0);
    k_res_ln<<<dim3(R), dim3(128), 0, stream>>>(bufF, bufB, hB, sp[l].g, sp[l].b, 1);
  }
  // sp_out = bufB

  // ---- mix + positional encoding ----
  k_mix_pe<<<dim3((unsigned)((RF + 255) / 256)), dim3(256), 0, stream>>>(bufB, bufA, bufC, hC, RF);

  // ---- temporal transformer ----
  const float iscale = 0.17677669529663687f;   // 1/sqrt(32)
  for (int l = 0; l < 2; ++l) {
    gemm(stream, hC, w_qkv[l], nullptr, qkvbf, tr[l].bqkv, RTOT, 384, FL, FL, 384, 384,
         0, 0, 0, 0, 0, 0, 1, 1, 0, 1.0f, 0.0f, 0);
    // scores = (q @ k^T)/sqrt(DH)  batches = 1600*4
    gemm(stream, qkvbf, qkvbf + 128, big, nullptr, nullptr, TT, TT, 32, 384, 384, TT,
         (long long)TT * 384, 32, (long long)TT * 384, 32,
         (long long)4 * TT * TT, (long long)TT * TT,
         4, BN * 4, 1, iscale, 0.0f, 0);
    k_softmax<<<dim3(BN * TT), dim3(64), 0, stream>>>(big, bigbf, TT);
    // o = att @ v
    gemm(stream, bigbf, qkvbf + 256, nullptr, hD, nullptr, TT, 32, TT, TT, 384, FL,
         (long long)4 * TT * TT, (long long)TT * TT,
         (long long)TT * 384, 32,
         (long long)TT * FL, 32,
         4, BN * 4, 0, 1.0f, 0.0f, 0);
    gemm(stream, hD, w_o[l], bufE, nullptr, tr[l].bo, RTOT, FL, FL, FL, FL, FL,
         0, 0, 0, 0, 0, 0, 1, 1, 0, 1.0f, 0.0f, 0);
    k_res_ln<<<dim3(R), dim3(128), 0, stream>>>(bufE, bufC, hC, tr[l].g1, tr[l].b1, 0);
    // ff = gelu(x@w1+bf1) @ w2 + bf2
    gemm(stream, hC, w_1[l], nullptr, bigbf, tr[l].bf1, RTOT, 512, FL, FL, 512, 512,
         0, 0, 0, 0, 0, 0, 1, 1, 0, 1.0f, 0.0f, 2);
    gemm(stream, bigbf, w_2[l], bufD, nullptr, tr[l].bf2, RTOT, FL, 512, 512, FL, FL,
         0, 0, 0, 0, 0, 0, 1, 1, 0, 1.0f, 0.0f, 0);
    k_res_ln<<<dim3(R), dim3(128), 0, stream>>>(bufD, bufC, hC, tr[l].g2, tr[l].b2, 0);
  }

  // ---- final transpose to [B,T,N,F] ----
  k_unpermute<<<dim3((unsigned)((RF + 255) / 256)), dim3(256), 0, stream>>>(bufC, (float*)d_out, RF);
  #undef IN
}
